// CrossAttnProcessor_52261162057998
// MI455X (gfx1250) — matrix operand order repaired and checked by reading, not compiled, not run on hardware
//
#include <hip/hip_runtime.h>

// ---------------------------------------------------------------------------
// Cross-attention (diffusers CrossAttnProcessor) for MI455X / gfx1250.
// B=4, N=M=2048, C=512, H=8, dh=64.  All matmuls in bf16 WMMA w/ f32 accum.
// Pipeline: proj Q/K/V (bf16) -> flash attention (online softmax) -> proj O.
// Round 3: fix async builtin arg types (int4 AS1* / int4 AS3* per diagnostic).
// ---------------------------------------------------------------------------

typedef __attribute__((ext_vector_type(16))) __bf16 v16bf;
typedef __attribute__((ext_vector_type(8)))  __bf16 v8bf;
typedef __attribute__((ext_vector_type(8)))  float  v8f;
typedef __attribute__((__vector_size__(4 * sizeof(int)))) int v4i;

constexpr int Bb = 4, Nn = 2048, Mm = 2048, Cc = 512, Hh = 8, Dh = 64;
constexpr int LD = 72; // LDS row stride (bf16 elems): 144B, 16B-aligned, bank-skewed

#define AS1 __attribute__((address_space(1)))
#define AS3 __attribute__((address_space(3)))

#if defined(__has_builtin)
#  if __has_builtin(__builtin_amdgcn_global_load_async_to_lds_b128)
#    define HAVE_ASYNC_LDS 1
#  else
#    define HAVE_ASYNC_LDS 0
#  endif
#  if __has_builtin(__builtin_amdgcn_s_wait_asynccnt)
#    define HAVE_WAIT_ASYNC 1
#  else
#    define HAVE_WAIT_ASYNC 0
#  endif
#  if __has_builtin(__builtin_amdgcn_update_dpp)
#    define HAVE_DPP 1
#  else
#    define HAVE_DPP 0
#  endif
#else
#  define HAVE_ASYNC_LDS 0
#  define HAVE_WAIT_ASYNC 0
#  define HAVE_DPP 0
#endif

__device__ __forceinline__ void wait_async0() {
#if HAVE_ASYNC_LDS
#  if HAVE_WAIT_ASYNC
  __builtin_amdgcn_s_wait_asynccnt(0);
#  else
  asm volatile("s_wait_asynccnt 0" ::: "memory");
#  endif
#endif
}

// ---- DPP16 butterfly reductions within each 16-lane half (softmax rows) ----
#if HAVE_DPP
template <int CTRL>
__device__ __forceinline__ float dpp_f(float x) {
  return __builtin_bit_cast(
      float, __builtin_amdgcn_update_dpp(0, __builtin_bit_cast(int, x), CTRL,
                                         0xF, 0xF, true));
}
__device__ __forceinline__ float redmax16(float v) {
  v = fmaxf(v, dpp_f<0xB1>(v));   // quad_perm(1,0,3,2) == xor1
  v = fmaxf(v, dpp_f<0x4E>(v));   // quad_perm(2,3,0,1) == xor2
  v = fmaxf(v, dpp_f<0x141>(v));  // row_half_mirror   == xor4 after above
  v = fmaxf(v, dpp_f<0x140>(v));  // row_mirror        == xor8 after above
  return v;
}
__device__ __forceinline__ float redsum16(float v) {
  v += dpp_f<0xB1>(v);
  v += dpp_f<0x4E>(v);
  v += dpp_f<0x141>(v);
  v += dpp_f<0x140>(v);
  return v;
}
#else
__device__ __forceinline__ float redmax16(float v) {
#pragma unroll
  for (int off = 1; off < 16; off <<= 1) v = fmaxf(v, __shfl_xor(v, off, 32));
  return v;
}
__device__ __forceinline__ float redsum16(float v) {
#pragma unroll
  for (int off = 1; off < 16; off <<= 1) v += __shfl_xor(v, off, 32);
  return v;
}
#endif

// A-matrix 16x32 bf16 fragment: lane holds row (lane%16);
// element e -> K = (e/8)*16 + halfsel*8 + (e%8)   (ISA 7.12.2 table)
__device__ __forceinline__ v16bf ld_fragA(const __bf16* row, int halfsel) {
  v8bf a = *(const v8bf*)(row + halfsel * 8);
  v8bf b = *(const v8bf*)(row + 16 + halfsel * 8);
  v16bf r;
#pragma unroll
  for (int i = 0; i < 8; ++i) { r[i] = a[i]; r[i + 8] = b[i]; }
  return r;
}

// B-matrix 32x16 bf16 fragment: lane holds column (lane%16);
// element e -> K = (e/8)*16 + halfsel*8 + (e%8)   (same as A)
__device__ __forceinline__ v16bf ld_fragB(const __bf16* col, int halfsel) {
  v8bf a = *(const v8bf*)(col + halfsel * 8);
  v8bf b = *(const v8bf*)(col + 16 + halfsel * 8);
  v16bf r;
#pragma unroll
  for (int i = 0; i < 8; ++i) { r[i] = a[i]; r[i + 8] = b[i]; }
  return r;
}

__device__ __forceinline__ v8f wmma_bf16(v16bf a, v16bf b, v8f c) {
  return __builtin_amdgcn_wmma_f32_16x16x32_bf16(false, a, false, b,
                                                 (short)0, c, false, false);
}

// ---------------------------------------------------------------------------
// Projection GEMM: Out = X[rows,512] * W[512,512]  (one 64-col tile == 1 head)
// MODE 0: X=f32 hidden,  Out = Q bf16 [B,H,N,64]
// MODE 1: X=f32 encoder, Out = K bf16 [B,H,M,64]
// MODE 2: X=f32 encoder, Out = V^T bf16 [B,H,64,M]
// MODE 3: X=bf16 attn,   Out = f32 [B*N,512] (+ bias)
// Block: 256 threads (8 waves), output tile 128 rows x 64 cols.
// ---------------------------------------------------------------------------
template <int MODE>
__global__ __launch_bounds__(256) void proj_kernel(const void* __restrict__ Xv,
                                                   const float* __restrict__ W,
                                                   const float* __restrict__ bias,
                                                   void* __restrict__ Out) {
  __shared__ __align__(16) __bf16 Xs[128 * LD];
  __shared__ __align__(16) __bf16 Wsh[64 * LD];
  const int tid = threadIdx.x, lane = tid & 31, wave = tid >> 5;
  const int halfsel = lane >> 4, l16 = lane & 15;
  const int rowbase = blockIdx.x * 128;
  const int colbase = blockIdx.y * 64;

  v8f acc[4];
#pragma unroll
  for (int t = 0; t < 4; ++t)
#pragma unroll
    for (int i = 0; i < 8; ++i) acc[t][i] = 0.f;

  for (int kc = 0; kc < Cc; kc += 64) {
    __syncthreads();
    // --- stage X tile (128 x 64) as bf16 ---
    if constexpr (MODE != 3) {
      const float* X = (const float*)Xv;
#pragma unroll
      for (int it = 0; it < 8; ++it) {
        int c = tid + it * 256;            // 2048 float4 chunks
        int r = c >> 4, k4 = (c & 15) * 4;
        float4 v = *(const float4*)(X + (size_t)(rowbase + r) * Cc + kc + k4);
        __bf16* dst = Xs + r * LD + k4;
        dst[0] = (__bf16)v.x; dst[1] = (__bf16)v.y;
        dst[2] = (__bf16)v.z; dst[3] = (__bf16)v.w;
      }
    } else {
      const __bf16* X = (const __bf16*)Xv;
#pragma unroll
      for (int it = 0; it < 4; ++it) {
        int c = tid + it * 256;            // 1024 chunks of 8 bf16
        int r = c >> 3, k8 = (c & 7) * 8;
        *(v8bf*)(Xs + r * LD + k8) =
            *(const v8bf*)(X + (size_t)(rowbase + r) * Cc + kc + k8);
      }
    }
    // --- stage W tile transposed: Wsh[n][k] = W[kc+k][colbase+n] ---
#pragma unroll
    for (int it = 0; it < 4; ++it) {
      int c = tid + it * 256;              // 64 k  x 16 float4-of-n
      int k = c >> 4, n4 = (c & 15) * 4;
      float4 v = *(const float4*)(W + (size_t)(kc + k) * Cc + colbase + n4);
      Wsh[(n4 + 0) * LD + k] = (__bf16)v.x;
      Wsh[(n4 + 1) * LD + k] = (__bf16)v.y;
      Wsh[(n4 + 2) * LD + k] = (__bf16)v.z;
      Wsh[(n4 + 3) * LD + k] = (__bf16)v.w;
    }
    __syncthreads();
    // --- compute: each wave does 16 rows x 64 cols, k-chunk of 64 ---
    const __bf16* arow = Xs + (wave * 16 + l16) * LD;
    v16bf a0 = ld_fragA(arow, halfsel);
    v16bf a1 = ld_fragA(arow + 32, halfsel);
#pragma unroll
    for (int t = 0; t < 4; ++t) {
      const __bf16* brow = Wsh + (t * 16 + l16) * LD;
      acc[t] = wmma_bf16(a0, ld_fragB(brow, halfsel), acc[t]);
      acc[t] = wmma_bf16(a1, ld_fragB(brow + 32, halfsel), acc[t]);
    }
  }

  // --- epilogue: C/D layout -> lane holds rows (halfsel*8 + r), col l16 ---
  const int h = blockIdx.y;
  const int row0 = rowbase + wave * 16 + halfsel * 8;
  if constexpr (MODE == 0 || MODE == 1) {
    constexpr int Seq = (MODE == 0) ? Nn : Mm;
    __bf16* O = (__bf16*)Out;
#pragma unroll
    for (int t = 0; t < 4; ++t) {
      int d = t * 16 + l16;
#pragma unroll
      for (int r = 0; r < 8; ++r) {
        int tok = row0 + r;
        int b = tok / Seq, s = tok % Seq;
        O[((size_t)(b * Hh + h) * Seq + s) * Dh + d] = (__bf16)acc[t][r];
      }
    }
  } else if constexpr (MODE == 2) {
    __bf16* O = (__bf16*)Out;
    int b = rowbase / Mm;          // 128-row block never crosses a batch
    int s0 = row0 - b * Mm;
#pragma unroll
    for (int t = 0; t < 4; ++t) {
      int d = t * 16 + l16;
      v8bf pk;
#pragma unroll
      for (int r = 0; r < 8; ++r) pk[r] = (__bf16)acc[t][r];
      *(v8bf*)(O + ((size_t)(b * Hh + h) * Dh + d) * Mm + s0) = pk;
    }
  } else {
    float* O = (float*)Out;
#pragma unroll
    for (int t = 0; t < 4; ++t) {
      int col = colbase + t * 16 + l16;
      float bv = bias[col];
#pragma unroll
      for (int r = 0; r < 8; ++r)
        O[(size_t)(row0 + r) * Cc + col] = acc[t][r] + bv;
    }
  }
}

// ---------------------------------------------------------------------------
// Flash attention: one block = 128 queries of one (b,h); 8 waves x 16 rows.
// Double-buffered K [64,64] / V^T [64,64] LDS tiles; async global->LDS when
// available. Online softmax in exp2 domain; P relaid D->A via per-wave LDS.
// ---------------------------------------------------------------------------
__global__ __launch_bounds__(256) void flash_kernel(const __bf16* __restrict__ Q,
                                                    const __bf16* __restrict__ K,
                                                    const __bf16* __restrict__ Vt,
                                                    __bf16* __restrict__ Oa) {
  __shared__ __align__(16) __bf16 Ks[2][64 * LD];
  __shared__ __align__(16) __bf16 Vs[2][64 * LD];     // V^T tile: [d][key]
  __shared__ __align__(16) __bf16 Ps[8][16 * LD];     // per-wave P scratch
  const int tid = threadIdx.x, lane = tid & 31, wave = tid >> 5;
  const int halfsel = lane >> 4, l16 = lane & 15;
  const int bh = blockIdx.y;
  const int b = bh >> 3, h = bh & 7;
  const int qbase = blockIdx.x * 128 + wave * 16;
  // softmax in exp2 domain: fold log2(e) into the 1/sqrt(dh) scale
  const float scale2 = 0.125f * 1.4426950408889634f;

  // Q fragments stay in registers for the whole K loop.
  const __bf16* qrow = Q + ((size_t)bh * Nn + qbase + l16) * Dh;
  v16bf qa0 = ld_fragA(qrow, halfsel);
  v16bf qa1 = ld_fragA(qrow + 32, halfsel);

  v8f o[4];
  float mr[8], lr[8];
#pragma unroll
  for (int t = 0; t < 4; ++t)
#pragma unroll
    for (int i = 0; i < 8; ++i) o[t][i] = 0.f;
#pragma unroll
  for (int r = 0; r < 8; ++r) { mr[r] = -3.0e38f; lr[r] = 0.f; }

  // Stage one 64-key tile (K rows + V^T rows) into LDS buffer `buf`.
  auto stage = [&](int kt, int buf) {
    const __bf16* Kg = K + ((size_t)bh * Mm + kt * 64) * Dh;
    const __bf16* Vg = Vt + (size_t)bh * Dh * Mm + (size_t)kt * 64;
#pragma unroll
    for (int it = 0; it < 2; ++it) {
      int c = tid + it * 256;          // 512 chunks of 8 bf16 each
      int r = c >> 3, k8 = (c & 7) * 8;
#if HAVE_ASYNC_LDS
      __builtin_amdgcn_global_load_async_to_lds_b128(
          (AS1 v4i*)(Kg + (size_t)r * Dh + k8),
          (AS3 v4i*)(&Ks[buf][r * LD + k8]), 0, 0);
      __builtin_amdgcn_global_load_async_to_lds_b128(
          (AS1 v4i*)(Vg + (size_t)r * Mm + k8),
          (AS3 v4i*)(&Vs[buf][r * LD + k8]), 0, 0);
#else
      *(v8bf*)(&Ks[buf][r * LD + k8]) = *(const v8bf*)(Kg + (size_t)r * Dh + k8);
      *(v8bf*)(&Vs[buf][r * LD + k8]) = *(const v8bf*)(Vg + (size_t)r * Mm + k8);
#endif
    }
  };

  stage(0, 0);
  wait_async0();
  __syncthreads();

  constexpr int NT = Mm / 64;
  for (int kt = 0; kt < NT; ++kt) {
    const int cur = kt & 1;
    if (kt + 1 < NT) stage(kt + 1, cur ^ 1);   // prefetch next tile

    // ---- S = Q * K^T for 4 key sub-tiles of 16 ----
    v8f s[4];
#pragma unroll
    for (int t = 0; t < 4; ++t) {
      v8f z;
#pragma unroll
      for (int i = 0; i < 8; ++i) z[i] = 0.f;
      const __bf16* krow = &Ks[cur][(t * 16 + l16) * LD]; // lane = key row
      z = wmma_bf16(qa0, ld_fragB(krow, halfsel), z);
      z = wmma_bf16(qa1, ld_fragB(krow + 32, halfsel), z);
      s[t] = z;
    }

    // ---- online softmax (exp2 domain): row stats across 4 tiles, 16 lanes ----
    float tmax[8];
#pragma unroll
    for (int r = 0; r < 8; ++r) {
      float v = s[0][r];
#pragma unroll
      for (int t = 1; t < 4; ++t) v = fmaxf(v, s[t][r]);
      tmax[r] = redmax16(v * scale2);
    }
    float corr[8], tsum[8];
#pragma unroll
    for (int r = 0; r < 8; ++r) {
      float mn = fmaxf(mr[r], tmax[r]);
      corr[r] = exp2f(mr[r] - mn);
      mr[r] = mn;
      tsum[r] = 0.f;
    }
#pragma unroll
    for (int t = 0; t < 4; ++t)
#pragma unroll
      for (int r = 0; r < 8; ++r) {
        float p = exp2f(s[t][r] * scale2 - mr[r]);
        s[t][r] = p;
        tsum[r] += p;
      }
#pragma unroll
    for (int r = 0; r < 8; ++r) {
      lr[r] = lr[r] * corr[r] + redsum16(tsum[r]);
    }
#pragma unroll
    for (int t = 0; t < 4; ++t)
#pragma unroll
      for (int r = 0; r < 8; ++r) o[t][r] *= corr[r];

    // ---- relayout P: D-fragment -> per-wave LDS -> A-fragment ----
    __bf16* Pw = Ps[wave];
#pragma unroll
    for (int t = 0; t < 4; ++t)
#pragma unroll
      for (int r = 0; r < 8; ++r)
        Pw[(halfsel * 8 + r) * LD + t * 16 + l16] = (__bf16)s[t][r];
    asm volatile("s_wait_dscnt 0" ::: "memory");   // wave-local RAW on LDS
    const __bf16* prow = Pw + (size_t)l16 * LD;
    v16bf pa0 = ld_fragA(prow, halfsel);
    v16bf pa1 = ld_fragA(prow + 32, halfsel);

    // ---- O += P * V : B-fragment = contiguous row of V^T ----
#pragma unroll
    for (int t = 0; t < 4; ++t) {
      const __bf16* vrow = &Vs[cur][(t * 16 + l16) * LD]; // lane = dh column
      o[t] = wmma_bf16(pa0, ld_fragB(vrow, halfsel), o[t]);
      o[t] = wmma_bf16(pa1, ld_fragB(vrow + 32, halfsel), o[t]);
    }

    wait_async0();      // our prefetch for tile kt+1 has landed
    __syncthreads();    // everyone done with Ks/Vs[cur] and prefetch visible
  }

  // ---- normalize and store attention output, heads concatenated ----
#pragma unroll
  for (int r = 0; r < 8; ++r) lr[r] = 1.0f / lr[r];
  const int row0 = qbase + halfsel * 8;
#pragma unroll
  for (int t = 0; t < 4; ++t) {
    int d = t * 16 + l16;
#pragma unroll
    for (int r = 0; r < 8; ++r)
      Oa[((size_t)b * Nn + row0 + r) * Cc + h * Dh + d] =
          (__bf16)(o[t][r] * lr[r]);
  }
}

// ---------------------------------------------------------------------------
extern "C" void kernel_launch(void* const* d_in, const int* in_sizes, int n_in,
                              void* d_out, int out_size, void* d_ws, size_t ws_size,
                              hipStream_t stream) {
  (void)in_sizes; (void)n_in; (void)out_size; (void)ws_size;
  const float* hidden  = (const float*)d_in[0];
  const float* encoder = (const float*)d_in[1];
  const float* Wq = (const float*)d_in[2];
  const float* Wk = (const float*)d_in[3];
  const float* Wv = (const float*)d_in[4];
  const float* Wo = (const float*)d_in[5];
  const float* bo = (const float*)d_in[6];

  char* ws = (char*)d_ws;
  const size_t sz = (size_t)Bb * Hh * Nn * Dh * sizeof(__bf16);  // 8 MB each
  __bf16* Qbf  = (__bf16*)(ws);
  __bf16* Kbf  = (__bf16*)(ws + sz);
  __bf16* Vtb  = (__bf16*)(ws + 2 * sz);
  __bf16* Attn = (__bf16*)(ws + 3 * sz);

  dim3 blk(256);
  dim3 gproj(Bb * Nn / 128, Cc / 64);     // 64 x 8 blocks (rows x heads/cols)
  hipLaunchKernelGGL((proj_kernel<0>), gproj, blk, 0, stream, (const void*)hidden,  Wq, nullptr, (void*)Qbf);
  hipLaunchKernelGGL((proj_kernel<1>), gproj, blk, 0, stream, (const void*)encoder, Wk, nullptr, (void*)Kbf);
  hipLaunchKernelGGL((proj_kernel<2>), gproj, blk, 0, stream, (const void*)encoder, Wv, nullptr, (void*)Vtb);

  dim3 gfa(Nn / 128, Bb * Hh);            // 16 x 32 blocks
  hipLaunchKernelGGL(flash_kernel, gfa, blk, 0, stream, Qbf, Kbf, Vtb, Attn);

  hipLaunchKernelGGL((proj_kernel<3>), gproj, blk, 0, stream, (const void*)Attn, Wo, bo, d_out);
}
